// LIIF_24575802868491
// MI455X (gfx1250) — matrix operand order repaired and checked by reading, not compiled, not run on hardware
//
#include <hip/hip_runtime.h>

typedef __attribute__((ext_vector_type(16))) _Float16 v16h;
typedef __attribute__((ext_vector_type(8)))  _Float16 v8h;
typedef __attribute__((ext_vector_type(8)))  float    v8f;
typedef int i32x4 __attribute__((vector_size(16)));   // matches builtin param type

union Frag { v16h v; v8h h[2]; };

// Detect gfx1250 async global->LDS path (ASYNCcnt-tracked, no VGPR roundtrip).
#if defined(__has_builtin)
#  if __has_builtin(__builtin_amdgcn_global_load_async_to_lds_b128) && \
      __has_builtin(__builtin_amdgcn_s_wait_asynccnt)
#    define USE_ASYNC_LDS 1
#  endif
#endif
#ifndef USE_ASYNC_LDS
#  define USE_ASYNC_LDS 0
#endif

// Problem constants (from reference setup_inputs)
constexpr int BSZ = 2, CCH = 64, HW = 64, QN = 65536;
constexpr int K0P = 608, KT0 = 19;             // layer-0 K padded to 19*32
constexpr int KTH = 8;                         // hidden layers: 8*32 = 256
constexpr int THREADS = 128;                   // 4 waves, 64 rows/block
constexpr int SLICE_H = 16 * 512;              // one K-slice: 16 ntiles x 512 halfs
// LDS layout (halfs)
constexpr int ACT_A_H = 64 * K0P;              // 38912
constexpr int ACT_B_H = 64 * 256;              // 16384
constexpr int SMEM_H  = ACT_A_H + ACT_B_H + 2 * SLICE_H; // 71680 halfs = 143360 B

// ---------------------------------------------------------------------------
// Weight conversion: f32 [kdim x ndim] row-major -> f16 WMMA B-fragments.
// Fragment (kt,nt) is 512 halfs, lane-major: element (lane,j) is
// k = kt*32 + (lane>>4)*8 + (j&7) + (j>>3)*16, n = nt*16 + (lane&15)   (ISA 7.12.2, B 32x16)
// ---------------------------------------------------------------------------
__global__ void convert_weights(const float* __restrict__ W, int kdim, int ndim,
                                int ntiles, int total, _Float16* __restrict__ dst) {
  int i = blockIdx.x * 256 + threadIdx.x;
  if (i >= total) return;
  int frag = i >> 9;
  int rem  = i & 511;
  int lane = rem >> 4;
  int j    = rem & 15;
  int kt   = frag / ntiles;
  int nt   = frag - kt * ntiles;
  int k = kt * 32 + (lane >> 4) * 8 + (j & 7) + (j >> 3) * 16;
  int n = nt * 16 + (lane & 15);
  float v = (k < kdim && n < ndim) ? W[(size_t)k * ndim + n] : 0.0f;
  dst[i] = (_Float16)v;
}

// ---------------------------------------------------------------------------
// Stage helpers: copy one 16 KB weight K-slice into an LDS buffer.
// ---------------------------------------------------------------------------
#if USE_ASYNC_LDS
__device__ __forceinline__ void stage_async(const _Float16* __restrict__ wslice,
                                            _Float16* buf) {
  const uint4* src = (const uint4*)wslice;
  uint4* dst = (uint4*)buf;
#pragma unroll
  for (int i = 0; i < 8; ++i)
    __builtin_amdgcn_global_load_async_to_lds_b128(
        (__attribute__((address_space(1))) i32x4*)(src + threadIdx.x + THREADS * i),
        (__attribute__((address_space(3))) i32x4*)(dst + threadIdx.x + THREADS * i),
        0, 0);
}
#endif

// ---------------------------------------------------------------------------
// Hidden layer: 16 rows/wave x 256 cols, K stepped by 32.
// Double-buffered LDS weight stage: prefetch slice kt+1 while computing kt;
// ONE barrier per K-step. Activations are wave-private -> no act barriers.
// ---------------------------------------------------------------------------
__device__ __forceinline__ void layer_hidden(
    const _Float16* __restrict__ actIn, int strideIn, int ktiles,
    const _Float16* __restrict__ wG, _Float16* bstage,
    const float* __restrict__ bias, _Float16* actOut) {
  const int lane = threadIdx.x & 31;
  const int wave = threadIdx.x >> 5;
  const v8f zero = {0.f, 0.f, 0.f, 0.f, 0.f, 0.f, 0.f, 0.f};
  v8f acc[16];
#pragma unroll
  for (int nt = 0; nt < 16; ++nt) acc[nt] = zero;

  // Preload slice 0 into buffer 0 (previous layer ended with a barrier).
#if USE_ASYNC_LDS
  stage_async(wG, bstage);
  __builtin_amdgcn_s_wait_asynccnt(0);
#else
  {
    const uint4* src = (const uint4*)wG;
    uint4* dst = (uint4*)bstage;
#pragma unroll
    for (int i = 0; i < 8; ++i)
      dst[threadIdx.x + THREADS * i] = src[threadIdx.x + THREADS * i];
  }
#endif
  __syncthreads();

  for (int kt = 0; kt < ktiles; ++kt) {
    _Float16* cur = bstage + (kt & 1) * SLICE_H;
    // ---- prefetch next slice (into the buffer last read at kt-1) ----
#if USE_ASYNC_LDS
    if (kt + 1 < ktiles)
      stage_async(wG + (size_t)(kt + 1) * SLICE_H, bstage + ((kt + 1) & 1) * SLICE_H);
#else
    uint4 wreg[8];
    if (kt + 1 < ktiles) {
      const uint4* src = (const uint4*)(wG + (size_t)(kt + 1) * SLICE_H);
#pragma unroll
      for (int i = 0; i < 8; ++i) wreg[i] = src[threadIdx.x + THREADS * i];
    }
#endif
    // ---- compute on current slice ----
    // A fragment (ISA 7.12.2, 16-bit A 16x32): lane m=lane&15,
    // lanes>=16 shift K by 8; two 16B chunks at K and K+16.
    Frag a;
    const _Float16* ap = actIn + (size_t)(wave * 16 + (lane & 15)) * strideIn
                       + kt * 32 + (lane >> 4) * 8;
    a.h[0] = *(const v8h*)ap;
    a.h[1] = *(const v8h*)(ap + 16);
#pragma unroll
    for (int nt = 0; nt < 16; ++nt) {
      Frag b;
      const _Float16* bp = cur + nt * 512 + lane * 16;
      b.h[0] = *(const v8h*)bp;
      b.h[1] = *(const v8h*)(bp + 8);
      acc[nt] = __builtin_amdgcn_wmma_f32_16x16x32_f16(
          false, a.v, false, b.v, (short)0, acc[nt], false, false);
    }
    // ---- publish prefetched slice, single barrier ----
#if USE_ASYNC_LDS
    __builtin_amdgcn_s_wait_asynccnt(0);
#else
    if (kt + 1 < ktiles) {
      uint4* dst = (uint4*)(bstage + ((kt + 1) & 1) * SLICE_H);
#pragma unroll
      for (int i = 0; i < 8; ++i) dst[threadIdx.x + THREADS * i] = wreg[i];
    }
#endif
    __syncthreads();
  }

  // Epilogue: D layout — VGPR r: lanes0-15 M=r, lanes16-31 M=r+8, N=lane&15.
  const int m0  = (lane >> 4) * 8;
  const int col = lane & 15;
#pragma unroll
  for (int nt = 0; nt < 16; ++nt) {
    float bv = bias[nt * 16 + col];
#pragma unroll
    for (int r = 0; r < 8; ++r) {
      float v = acc[nt][r] + bv;
      v = v > 0.f ? v : 0.f;                 // ReLU
      actOut[(size_t)(wave * 16 + m0 + r) * 256 + nt * 16 + col] = (_Float16)v;
    }
  }
}

// ---------------------------------------------------------------------------
// Final layer: 256 -> 3 (padded to 16 cols). B fragments read directly from
// global (1 KB/K-step, L2 broadcast) -> completely barrier-free.
// ---------------------------------------------------------------------------
__device__ __forceinline__ void layer_final(
    const _Float16* __restrict__ actIn, const _Float16* __restrict__ wG,
    const float* __restrict__ bias, float* __restrict__ preds, long rowBase) {
  const int lane = threadIdx.x & 31;
  const int wave = threadIdx.x >> 5;
  v8f acc = {0.f, 0.f, 0.f, 0.f, 0.f, 0.f, 0.f, 0.f};
#pragma unroll
  for (int kt = 0; kt < KTH; ++kt) {
    Frag a;
    const _Float16* ap = actIn + (size_t)(wave * 16 + (lane & 15)) * 256
                       + kt * 32 + (lane >> 4) * 8;
    a.h[0] = *(const v8h*)ap;
    a.h[1] = *(const v8h*)(ap + 16);
    Frag b;
    const _Float16* bp = wG + (size_t)kt * 512 + lane * 16;
    b.h[0] = *(const v8h*)bp;
    b.h[1] = *(const v8h*)(bp + 8);
    acc = __builtin_amdgcn_wmma_f32_16x16x32_f16(
        false, a.v, false, b.v, (short)0, acc, false, false);
  }
  const int m0 = (lane >> 4) * 8;
  const int col = lane & 15;
  if (col < 3) {
    float bv = bias[col];
#pragma unroll
    for (int r = 0; r < 8; ++r) {
      long row = rowBase + wave * 16 + m0 + r;
      preds[row * 3 + col] = acc[r] + bv;
    }
  }
}

// ---------------------------------------------------------------------------
// Fused MLP kernel: gather 580-dim inputs for 64 rows, run 5 layers in LDS.
// Row id = ((shift*2 + b)<<16) + q.
// ---------------------------------------------------------------------------
__global__ void __launch_bounds__(THREADS)
mlp_kernel(const float* __restrict__ feat, const float* __restrict__ coord,
           const float* __restrict__ cell,
           const float* __restrict__ b0, const float* __restrict__ b1,
           const float* __restrict__ b2, const float* __restrict__ b3,
           const float* __restrict__ b4,
           const _Float16* __restrict__ w0, const _Float16* __restrict__ w1,
           const _Float16* __restrict__ w2, const _Float16* __restrict__ w3,
           const _Float16* __restrict__ w4, float* __restrict__ preds) {
  extern __shared__ _Float16 smem[];
  _Float16* actA = smem;                         // 64 x 608 (layer0 in)
  _Float16* actB = smem + ACT_A_H;               // 64 x 256 (ping-pong)
  _Float16* bstg = smem + ACT_A_H + ACT_B_H;     // 2 x 16 KB weight stage

  // ---- gather: 2 threads per row (part 0/1 handle 32 channels each) ----
  {
    const int t = threadIdx.x;
    const int rl = t >> 1, part = t & 1;
    const long rg = (long)blockIdx.x * 64 + rl;
    const int s   = (int)(rg >> 17);
    const int rem = (int)(rg & 131071);
    const int b   = rem >> 16, q = rem & 65535;
    const float c0 = coord[((size_t)b * QN + q) * 2 + 0];
    const float c1 = coord[((size_t)b * QN + q) * 2 + 1];
    const float vx = (s < 2) ? -1.f : 1.f;
    const float vy = (s & 1) ? 1.f : -1.f;
    const float r = 1.f / 64.f;
    float cy = fminf(fmaxf(c0 + vx * r + 1e-6f, -1.f + 1e-6f), 1.f - 1e-6f);
    float cx = fminf(fmaxf(c1 + vy * r + 1e-6f, -1.f + 1e-6f), 1.f - 1e-6f);
    const int iy = (int)fminf(fmaxf(rintf((cy + 1.f) * 32.f - 0.5f), 0.f), 63.f);
    const int ix = (int)fminf(fmaxf(rintf((cx + 1.f) * 32.f - 0.5f), 0.f), 63.f);
    int off[9];
#pragma unroll
    for (int tap = 0; tap < 9; ++tap) {
      int y = iy + tap / 3 - 1, x = ix + tap % 3 - 1;
      off[tap] = (y >= 0 && y < HW && x >= 0 && x < HW) ? (y * HW + x) : -1;
    }
    const float* fb = feat + (size_t)b * CCH * (HW * HW);
    _Float16* dst = actA + (size_t)rl * K0P;
    for (int c = part * 32; c < part * 32 + 32; ++c) {
      const float* fc = fb + (size_t)c * (HW * HW);
#pragma unroll
      for (int tap = 0; tap < 9; ++tap) {
        float v = (off[tap] >= 0) ? fc[off[tap]] : 0.f;
        dst[c * 9 + tap] = (_Float16)v;
      }
    }
    if (part == 0) {
      float qcy = (2.f * iy + 1.f) * r - 1.f;
      float qcx = (2.f * ix + 1.f) * r - 1.f;
      dst[576] = (_Float16)((c0 - qcy) * 64.f);   // rel_y
      dst[577] = (_Float16)((c1 - qcx) * 64.f);   // rel_x
    } else {
      dst[578] = (_Float16)(cell[((size_t)b * QN + q) * 2 + 0] * 64.f);
      dst[579] = (_Float16)(cell[((size_t)b * QN + q) * 2 + 1] * 64.f);
    }
#pragma unroll
    for (int k = 0; k < 14; ++k)
      dst[580 + part * 14 + k] = (_Float16)0.f;   // K padding
  }

  layer_hidden(actA, K0P, KT0, w0, bstg, b0, actB);
  layer_hidden(actB, 256, KTH, w1, bstg, b1, actA);
  layer_hidden(actA, 256, KTH, w2, bstg, b2, actB);
  layer_hidden(actB, 256, KTH, w3, bstg, b3, actA);
  layer_final(actA, w4, b4, preds, (long)blockIdx.x * 64);
}

// ---------------------------------------------------------------------------
// Ensemble: slot self-attention weighting of the 4 shift predictions.
// ---------------------------------------------------------------------------
__global__ void __launch_bounds__(256)
ensemble_kernel(const float* __restrict__ feat, const float* __restrict__ coord,
                const float* __restrict__ preds, float* __restrict__ out) {
  int t = blockIdx.x * 256 + threadIdx.x;
  if (t >= BSZ * QN) return;
  const int b = t >> 16, q = t & 65535;
  const float c0 = coord[((size_t)b * QN + q) * 2 + 0];
  const float c1 = coord[((size_t)b * QN + q) * 2 + 1];
  const float r = 1.f / 64.f;
  int off[4];
#pragma unroll
  for (int s = 0; s < 4; ++s) {
    float vx = (s < 2) ? -1.f : 1.f;
    float vy = (s & 1) ? 1.f : -1.f;
    float cy = fminf(fmaxf(c0 + vx * r + 1e-6f, -1.f + 1e-6f), 1.f - 1e-6f);
    float cx = fminf(fmaxf(c1 + vy * r + 1e-6f, -1.f + 1e-6f), 1.f - 1e-6f);
    int iy = (int)fminf(fmaxf(rintf((cy + 1.f) * 32.f - 0.5f), 0.f), 63.f);
    int ix = (int)fminf(fmaxf(rintf((cx + 1.f) * 32.f - 0.5f), 0.f), 63.f);
    off[s] = iy * HW + ix;
  }
  const float* fb = feat + (size_t)b * CCH * (HW * HW);
  float n0 = 0, n1 = 0, n2 = 0, n3 = 0;
  float d01 = 0, d02 = 0, d03 = 0, d12 = 0, d13 = 0, d23 = 0;
  for (int c = 0; c < 64; ++c) {               // SLOT_DIM == C == 64
    const float* fc = fb + (size_t)c * (HW * HW);
    float v0 = fc[off[0]], v1 = fc[off[1]], v2 = fc[off[2]], v3 = fc[off[3]];
    n0 += v0 * v0; n1 += v1 * v1; n2 += v2 * v2; n3 += v3 * v3;
    d01 += v0 * v1; d02 += v0 * v2; d03 += v0 * v3;
    d12 += v1 * v2; d13 += v1 * v3; d23 += v2 * v3;
  }
  float i0 = 1.f / sqrtf(n0), i1 = 1.f / sqrtf(n1);
  float i2 = 1.f / sqrtf(n2), i3 = 1.f / sqrtf(n3);
  const float sc = 1.f / 64.f;                 // / SLOT_DIM
  float a01 = d01 * i0 * i1 * sc, a02 = d02 * i0 * i2 * sc, a03 = d03 * i0 * i3 * sc;
  float a12 = d12 * i1 * i2 * sc, a13 = d13 * i1 * i3 * sc, a23 = d23 * i2 * i3 * sc;
  float w0 = (a01 + a02 + a03) * 0.25f;
  float w1 = (a01 + a12 + a13) * 0.25f;
  float w2 = (a02 + a12 + a23) * 0.25f;
  float w3 = (a03 + a13 + a23) * 0.25f;
  float inv = 1.f / (w0 + w1 + w2 + w3);
  w0 *= inv; w1 *= inv; w2 *= inv; w3 *= inv;
#pragma unroll
  for (int k = 0; k < 3; ++k) {
    float p0 = preds[((size_t)(0 << 17) + t) * 3 + k];
    float p1 = preds[((size_t)(1 << 17) + t) * 3 + k];
    float p2 = preds[((size_t)(2 << 17) + t) * 3 + k];
    float p3 = preds[((size_t)(3 << 17) + t) * 3 + k];
    out[(size_t)t * 3 + k] = p0 * w0 + p1 * w1 + p2 * w2 + p3 * w3;
  }
}

// ---------------------------------------------------------------------------
extern "C" void kernel_launch(void* const* d_in, const int* in_sizes, int n_in,
                              void* d_out, int out_size, void* d_ws, size_t ws_size,
                              hipStream_t stream) {
  (void)in_sizes; (void)n_in; (void)out_size; (void)ws_size;
  const float* feat  = (const float*)d_in[0];
  const float* coord = (const float*)d_in[1];
  const float* cell  = (const float*)d_in[2];
  const float* W0 = (const float*)d_in[3];  const float* b0 = (const float*)d_in[4];
  const float* W1 = (const float*)d_in[5];  const float* b1 = (const float*)d_in[6];
  const float* W2 = (const float*)d_in[7];  const float* b2 = (const float*)d_in[8];
  const float* W3 = (const float*)d_in[9];  const float* b3 = (const float*)d_in[10];
  const float* W4 = (const float*)d_in[11]; const float* b4 = (const float*)d_in[12];
  float* out = (float*)d_out;

  // Workspace: f16 weight fragments + f32 preds
  _Float16* w0h = (_Float16*)d_ws;           // 19*16*512 halfs
  _Float16* w1h = w0h + 19 * 16 * 512;
  _Float16* w2h = w1h + 8 * 16 * 512;
  _Float16* w3h = w2h + 8 * 16 * 512;
  _Float16* w4h = w3h + 8 * 16 * 512;
  float* preds  = (float*)(w4h + 8 * 512);   // 524288 * 3 floats

  {
    int total = 19 * 16 * 512;
    convert_weights<<<(total + 255) / 256, 256, 0, stream>>>(W0, 580, 256, 16, total, w0h);
    total = 8 * 16 * 512;
    convert_weights<<<(total + 255) / 256, 256, 0, stream>>>(W1, 256, 256, 16, total, w1h);
    convert_weights<<<(total + 255) / 256, 256, 0, stream>>>(W2, 256, 256, 16, total, w2h);
    convert_weights<<<(total + 255) / 256, 256, 0, stream>>>(W3, 256, 256, 16, total, w3h);
    total = 8 * 1 * 512;
    convert_weights<<<(total + 255) / 256, 256, 0, stream>>>(W4, 256, 3, 1, total, w4h);
  }

  const size_t smemBytes = (size_t)SMEM_H * sizeof(_Float16);  // 143360 B
  mlp_kernel<<<8192, THREADS, smemBytes, stream>>>(
      feat, coord, cell, b0, b1, b2, b3, b4, w0h, w1h, w2h, w3h, w4h, preds);

  ensemble_kernel<<<512, 256, 0, stream>>>(feat, coord, preds, out);
}
